// ODEJump_73770358276752
// MI455X (gfx1250) — matrix operand order repaired and checked by reading, not compiled, not run on hardware
//
#include <hip/hip_runtime.h>
#include <math.h>
#include <stdint.h>

// Problem dims
#define Bn 128
#define Tn 512
#define Cn 64
#define Hn 256
#define H4 1024
#define G3 768

typedef __bf16 bf16;
typedef __attribute__((ext_vector_type(16))) __bf16 v16bf;
typedef __attribute__((ext_vector_type(8)))  __bf16 v8bf;
typedef __attribute__((ext_vector_type(8)))  float  v8f;
typedef __attribute__((ext_vector_type(4))) unsigned int u32x4;
typedef __attribute__((ext_vector_type(4))) int i32x4;
typedef __attribute__((ext_vector_type(8))) int i32x8;

// Staged B-tile row pitch in LDS: 32 bf16 data (64B) + 16B TDM pad = 40 elements
#define WPITCH 40

// ---------------- WMMA fragment helpers (wave32, 16x16x32 bf16) ----------------
__device__ __forceinline__ v16bf fragA(const bf16* base, int ld, int kbase, int lane) {
  const int r = lane & 15, kg = lane >> 4;
  const bf16* p = base + (size_t)r * ld + kbase + 16 * kg;
  v8bf lo = *(const v8bf*)(p);
  v8bf hi = *(const v8bf*)(p + 8);
  return __builtin_shufflevector(lo, hi, 0,1,2,3,4,5,6,7,8,9,10,11,12,13,14,15);
}
// B fragment from global, weight pre-transposed [N][K] row-major
__device__ __forceinline__ v16bf fragB(const bf16* wt, int ld, int n0, int kbase, int lane) {
  const bf16* p = wt + (size_t)(n0 + (lane & 15)) * ld + kbase + 16 * (lane >> 4);
  v8bf lo = *(const v8bf*)(p);
  v8bf hi = *(const v8bf*)(p + 8);
  return __builtin_shufflevector(lo, hi, 0,1,2,3,4,5,6,7,8,9,10,11,12,13,14,15);
}
// B fragment from TDM-staged LDS slab (rows x 32 K, pitch WPITCH)
__device__ __forceinline__ v16bf fragB_lds(const bf16* sb, int nloc, int lane) {
  const bf16* p = sb + (nloc + (lane & 15)) * WPITCH + 16 * (lane >> 4);
  v8bf lo = *(const v8bf*)(p);
  v8bf hi = *(const v8bf*)(p + 8);
  return __builtin_shufflevector(lo, hi, 0,1,2,3,4,5,6,7,8,9,10,11,12,13,14,15);
}
__device__ __forceinline__ v8f wmma_bf16(v16bf a, v16bf b, v8f c) {
  return __builtin_amdgcn_wmma_f32_16x16x32_bf16(false, a, false, b, (short)0, c, false, false);
}
__device__ __forceinline__ float sigm(float x) { return 1.f / (1.f + __expf(-x)); }
__device__ __forceinline__ float wred(float v) {
#pragma unroll
  for (int m = 16; m >= 1; m >>= 1) v += __shfl_xor(v, m, 32);
  return v;
}

// ---------------- TDM: async 2D bf16 tile load global -> LDS (CDNA5 ISA ch.8) ----------------
// group0 = {count, lds_addr, global_addr, type=2}; group1 = dims/strides/pad.
__device__ __forceinline__ void tdm_load_2d(const bf16* gsrc, unsigned lds_off,
                                            int rows, int cols, int row_stride_elems,
                                            bool pad) {
  unsigned long long ga = (unsigned long long)(uintptr_t)gsrc;
  u32x4 g0;
  g0[0] = 1u;                                                // count=1, user desc
  g0[1] = lds_off;                                           // lds_addr (bytes)
  g0[2] = (unsigned)ga;                                      // global_addr[31:0]
  g0[3] = (unsigned)((ga >> 32) & 0x01FFFFFFu) | (2u << 30); // addr[56:32] | type=2
  i32x8 g1;
  // word0: data_size=1 (2B); optionally pad_enable=1, pad_interval=3 (16 dw = 64B row),
  //        pad_amount=3 (4 dwords = 16B) -> 80B LDS row pitch
  g1[0] = (1 << 16) | (pad ? ((1 << 20) | (3 << 22) | (3 << 25)) : 0);
  g1[1] = (int)((unsigned)cols << 16);      // tensor_dim0[15:0] at bits[63:48]
  g1[2] = (int)((unsigned)rows << 16);      // tensor_dim1[15:0] at bits[95:80]
  g1[3] = (int)((unsigned)cols << 16);      // tile_dim0 at bits[127:112]
  g1[4] = rows;                             // tile_dim1 at bits[143:128]
  g1[5] = row_stride_elems;                 // tensor_dim0_stride[31:0]
  g1[6] = 0;
  g1[7] = 0;
  i32x4 z4 = {0, 0, 0, 0};
#if __clang_major__ >= 23
  i32x8 z8 = {0, 0, 0, 0, 0, 0, 0, 0};
  __builtin_amdgcn_tensor_load_to_lds(g0, g1, z4, z4, z8, 0);
#else
  __builtin_amdgcn_tensor_load_to_lds(g0, g1, z4, z4, 0);
#endif
}

// ---------------- prep: f32 [K][N] -> bf16 transposed [N][K] ----------------
__global__ void k_tr(const float* __restrict__ in, bf16* __restrict__ out, int K, int N) {
  int idx = blockIdx.x * blockDim.x + threadIdx.x;
  if (idx < K * N) {
    int n = idx / K, k = idx - n * K;
    out[idx] = (bf16)in[(size_t)k * N + n];
  }
}

// ---------------- encoder: e = relu([x,mask]@W1+b1)@W2+b2 (bf16 out) ----------------
#define LDE (2 * Cn + 8)
#define LDM (H4 + 8)
__global__ __launch_bounds__(512, 1)
void k_encoder(const float* __restrict__ x, const float* __restrict__ msk,
               const bf16* __restrict__ W1t, const float* __restrict__ b1,
               const bf16* __restrict__ W2t, const float* __restrict__ b2,
               bf16* __restrict__ e) {
  __shared__ __align__(16) bf16 zA[16][LDE];
  __shared__ __align__(16) bf16 mid[16][LDM];
  const int tid = threadIdx.x, lane = tid & 31, wid = tid >> 5;
  const size_t r0 = (size_t)blockIdx.x * 16;  // flattened row over B*T

  for (int idx = tid; idx < 16 * Cn; idx += 512) {
    int m = idx >> 6, c = idx & 63;
    zA[m][c]      = (bf16)x[(r0 + m) * Cn + c];
    zA[m][Cn + c] = (bf16)msk[(r0 + m) * Cn + c];
  }
  __syncthreads();
  const int mb = (lane >> 4) * 8;
  {
    v8f a0 = {}, a1 = {}, a2 = {}, a3 = {};
#pragma unroll
    for (int kk = 0; kk < 4; ++kk) {
      v16bf av = fragA(&zA[0][0], LDE, kk * 32, lane);
      v16bf bv;
      bv = fragB(W1t, 2 * Cn, wid * 16 + 0,   kk * 32, lane); a0 = wmma_bf16(av, bv, a0);
      bv = fragB(W1t, 2 * Cn, wid * 16 + 256, kk * 32, lane); a1 = wmma_bf16(av, bv, a1);
      bv = fragB(W1t, 2 * Cn, wid * 16 + 512, kk * 32, lane); a2 = wmma_bf16(av, bv, a2);
      bv = fragB(W1t, 2 * Cn, wid * 16 + 768, kk * 32, lane); a3 = wmma_bf16(av, bv, a3);
    }
#pragma unroll
    for (int q = 0; q < 4; ++q) {
      v8f acc = (q == 0) ? a0 : (q == 1) ? a1 : (q == 2) ? a2 : a3;
      int n = wid * 16 + q * 256 + (lane & 15);
      float bias = b1[n];
#pragma unroll
      for (int i = 0; i < 8; ++i) {
        float v = acc[i] + bias;
        mid[mb + i][n] = (bf16)(v > 0.f ? v : 0.f);
      }
    }
  }
  __syncthreads();
  {
    v8f acc = {};
#pragma unroll 4
    for (int kk = 0; kk < 32; ++kk) {
      v16bf av = fragA(&mid[0][0], LDM, kk * 32, lane);
      v16bf bv = fragB(W2t, H4, wid * 16, kk * 32, lane);
      acc = wmma_bf16(av, bv, acc);
    }
    int n = wid * 16 + (lane & 15);
    float bias = b2[n];
#pragma unroll
    for (int i = 0; i < 8; ++i)
      e[(r0 + mb + i) * Hn + n] = (bf16)(acc[i] + bias);
  }
}

// ---------------- persistent scan: 8 workgroups x 16 batch rows, full T loop ----------------
#define LDZ (H4 + 8)
__global__ __launch_bounds__(512, 1)
void k_scan(const bf16* __restrict__ e, const float* __restrict__ ts,
            const bf16* __restrict__ W1t, const float* __restrict__ b1,
            const bf16* __restrict__ W2t, const float* __restrict__ b2,
            const bf16* __restrict__ Wiht, const bf16* __restrict__ Whht,
            const float* __restrict__ bih, const float* __restrict__ bhh,
            const float* __restrict__ lnog, const float* __restrict__ lnob,
            const float* __restrict__ lngg, const float* __restrict__ lngb,
            float* __restrict__ out) {
  __shared__ __align__(16) bf16 zb[16][LDZ];          // z = [h, xi, xl, sl] bf16
  __shared__ __align__(16) bf16 gs[16][LDZ];          // silu(z@W1+b1) bf16 / f32 scratch
  __shared__ __align__(16) bf16 ebl[2][16][Hn];       // TDM double buffer of e[:,t] tiles
  __shared__ __align__(16) bf16 wstage[16][2][64][WPITCH];  // per-wave TDM B-tile staging
  __shared__ float hcur[16][Hn];
  __shared__ float hacc[16][Hn];                      // RK4 accumulator, then h_ode
  __shared__ float dts[16];
  float* hgru = (float*)&gs[0][0];                    // 16*Hn f32 reuse inside gs

  const int tid = threadIdx.x, lane = tid & 31, wid = tid >> 5;
  const int rb = blockIdx.x * 16;
  const bf16* ebase = e + (size_t)rb * Tn * Hn;

  auto prefetch_e = [&](int t, int buf) {  // async TDM tile: 16 rows x Hn, stride T*H
    if (wid == 0)
      tdm_load_2d(ebase + (size_t)t * Hn, (unsigned)(uintptr_t)&ebl[buf][0][0],
                  16, Hn, Tn * Hn, false);
  };
  // per-wave staged B slabs (double buffered, one K-chunk ahead)
  auto stage_w1 = [&](int kk, int buf) {  // 64 rows (n = wid*64..) x 32 K
    tdm_load_2d(W1t + (size_t)(wid * 64) * H4 + kk * 32,
                (unsigned)(uintptr_t)&wstage[wid][buf][0][0], 64, 32, H4, true);
  };
  auto stage_w2 = [&](int kk, int buf) {  // 16 rows (n = wid*16..) x 32 K
    tdm_load_2d(W2t + (size_t)(wid * 16) * H4 + kk * 32,
                (unsigned)(uintptr_t)&wstage[wid][buf][0][0], 16, 32, H4, true);
  };

  // one RK4 stage: k = silu(z@W1+b1)@W2+b2 ; hacc += wk*dt*k ; z.h = h + cs*dt*k
  auto eval = [&](float wk, float cs, bool first) {
    const int mb = (lane >> 4) * 8;
    {  // GEMM1: per-wave 64 contiguous N-cols, B slabs DMA'd one chunk ahead
      if (first) stage_w1(0, 0);  // otherwise chunk0 was issued at end of previous phase
      v8f a0 = {}, a1 = {}, a2 = {}, a3 = {};
#pragma unroll 2
      for (int kk = 0; kk < 32; ++kk) {
        stage_w1((kk + 1) & 31, (kk + 1) & 1);       // prefetch next chunk (wraps)
        __builtin_amdgcn_s_wait_tensorcnt(1);        // chunk kk resident
        v16bf av = fragA(&zb[0][0], LDZ, kk * 32, lane);
        const bf16* sb = &wstage[wid][kk & 1][0][0];
        v16bf b0 = fragB_lds(sb, 0,  lane);
        v16bf b1v = fragB_lds(sb, 16, lane);
        v16bf b2v = fragB_lds(sb, 32, lane);
        v16bf b3v = fragB_lds(sb, 48, lane);
        a0 = wmma_bf16(av, b0, a0);
        a1 = wmma_bf16(av, b1v, a1);
        a2 = wmma_bf16(av, b2v, a2);
        a3 = wmma_bf16(av, b3v, a3);
      }
      __builtin_amdgcn_s_wait_tensorcnt(0);          // drain wrap-around issue
#pragma unroll
      for (int q = 0; q < 4; ++q) {
        v8f acc = (q == 0) ? a0 : (q == 1) ? a1 : (q == 2) ? a2 : a3;
        int n = wid * 64 + q * 16 + (lane & 15);
        float bias = b1[n];
#pragma unroll
        for (int i = 0; i < 8; ++i) {
          float v = acc[i] + bias;
          gs[mb + i][n] = (bf16)(v * sigm(v));  // SiLU
        }
      }
    }
    stage_w2(0, 0);  // overlap barrier latency with first W2 chunk DMA
    __syncthreads();
    {  // GEMM2: 1 N-tile per wave, B slabs DMA'd one chunk ahead
      v8f acc = {};
#pragma unroll 2
      for (int kk = 0; kk < 32; ++kk) {
        stage_w2((kk + 1) & 31, (kk + 1) & 1);
        __builtin_amdgcn_s_wait_tensorcnt(1);
        v16bf av = fragA(&gs[0][0], LDZ, kk * 32, lane);
        v16bf bv = fragB_lds(&wstage[wid][kk & 1][0][0], 0, lane);
        acc = wmma_bf16(av, bv, acc);
      }
      __builtin_amdgcn_s_wait_tensorcnt(0);
      int n = wid * 16 + (lane & 15);
      float bias = b2[n];
#pragma unroll
      for (int i = 0; i < 8; ++i) {
        int m = mb + i;
        float kv = acc[i] + bias;
        float dtm = dts[m];
        hacc[m][n] += wk * dtm * kv;
        if (cs >= 0.f) zb[m][n] = (bf16)(hcur[m][n] + cs * dtm * kv);
      }
    }
    if (cs >= 0.f) stage_w1(0, 0);  // next eval's first W1 chunk, overlapping the barrier
    __syncthreads();
  };

  // GRU(xi = zb[:,H:2H], h = hacc/zb[:,0:H]) + LayerNorm -> hcur, out[:,t]
  auto gru = [&](int t) {
    const int mb = (lane >> 4) * 8;
    v8f ar = {}, az = {}, an = {}, br = {}, bz = {}, bn_ = {};
#pragma unroll
    for (int kk = 0; kk < 8; ++kk) {
      v16bf axi = fragA(&zb[0][Hn], LDZ, kk * 32, lane);
      v16bf ah  = fragA(&zb[0][0],  LDZ, kk * 32, lane);
      v16bf bv;
      bv = fragB(Wiht, Hn, wid * 16,          kk * 32, lane); ar  = wmma_bf16(axi, bv, ar);
      bv = fragB(Wiht, Hn, Hn + wid * 16,     kk * 32, lane); az  = wmma_bf16(axi, bv, az);
      bv = fragB(Wiht, Hn, 2 * Hn + wid * 16, kk * 32, lane); an  = wmma_bf16(axi, bv, an);
      bv = fragB(Whht, Hn, wid * 16,          kk * 32, lane); br  = wmma_bf16(ah,  bv, br);
      bv = fragB(Whht, Hn, Hn + wid * 16,     kk * 32, lane); bz  = wmma_bf16(ah,  bv, bz);
      bv = fragB(Whht, Hn, 2 * Hn + wid * 16, kk * 32, lane); bn_ = wmma_bf16(ah,  bv, bn_);
    }
    int n = wid * 16 + (lane & 15);
    float bir = bih[n], biz = bih[Hn + n], bin = bih[2 * Hn + n];
    float bhr = bhh[n], bhz = bhh[Hn + n], bhn = bhh[2 * Hn + n];
#pragma unroll
    for (int i = 0; i < 8; ++i) {
      int m = mb + i;
      float r  = sigm(ar[i] + bir + br[i] + bhr);
      float zg = sigm(az[i] + biz + bz[i] + bhz);
      float nn = tanhf(an[i] + bin + r * (bn_[i] + bhn));
      float hp = hacc[m][n];
      hgru[m * Hn + n] = (1.f - zg) * nn + zg * hp;
    }
    __syncthreads();
    {
      int row = wid;
      float v[8], s = 0.f, s2 = 0.f;
#pragma unroll
      for (int i = 0; i < 8; ++i) {
        v[i] = hgru[row * Hn + i * 32 + lane];
        s += v[i]; s2 += v[i] * v[i];
      }
      s = wred(s); s2 = wred(s2);
      float mean = s * (1.f / Hn);
      float rstd = rsqrtf(s2 * (1.f / Hn) - mean * mean + 1e-5f);
#pragma unroll
      for (int i = 0; i < 8; ++i) {
        int c = i * 32 + lane;
        float hv = (v[i] - mean) * rstd * lngg[c] + lngb[c];
        hcur[row][c] = hv;
        out[((size_t)(rb + row) * Tn + t) * Hn + c] = hv;
      }
    }
    __syncthreads();
  };

  // ---- prime the TDM double buffer with e[:,0], e[:,1] ----
  prefetch_e(0, 0);
  prefetch_e(1, 1);
  if (wid == 0) __builtin_amdgcn_s_wait_tensorcnt(0);
  __syncthreads();

  // ---- t = 0: h0 = LN(GRU(e[:,0], 0)) ----
  for (int idx = tid; idx < 16 * Hn; idx += 512) {
    int m = idx >> 8, c = idx & (Hn - 1);
    zb[m][Hn + c] = ebl[0][m][c];
    zb[m][c] = (bf16)0.f;
    hacc[m][c] = 0.f;
  }
  __syncthreads();
  gru(0);

  // ---- t = 1 .. T-1 ----
  for (int t = 1; t < Tn; ++t) {
    if (wid == 0) __builtin_amdgcn_s_wait_tensorcnt(0);  // e[:,t] tile arrived
    if (tid < 16) dts[tid] = ts[(rb + tid) * Tn + t] - ts[(rb + tid) * Tn + t - 1];
    __syncthreads();
    const int bi = t & 1, bl = (t - 1) & 1;
    for (int idx = tid; idx < 16 * Hn; idx += 512) {
      int m = idx >> 8, c = idx & (Hn - 1);
      float xi = (float)ebl[bi][m][c];
      float xl = (float)ebl[bl][m][c];
      zb[m][Hn + c]     = (bf16)xi;
      zb[m][2 * Hn + c] = (bf16)xl;
      zb[m][3 * Hn + c] = (bf16)((xi - xl) / dts[m]);
      zb[m][c]          = (bf16)hcur[m][c];
      hacc[m][c]        = 0.f;
    }
    __syncthreads();
    if (t + 1 < Tn) prefetch_e(t + 1, (t + 1) & 1);  // overlap TDM with RK4 evals

    eval(1.f / 6.f, 0.5f, true);
    eval(2.f / 6.f, 0.5f, false);
    eval(2.f / 6.f, 1.0f, false);
    eval(1.f / 6.f, -1.f, false);

    {  // h = LN_ode(h + hacc) -> hacc (f32 for GRU) and zb[:,0:H] (bf16 A-tile)
      int row = wid;
      float v[8], s = 0.f, s2 = 0.f;
#pragma unroll
      for (int i = 0; i < 8; ++i) {
        float hv = hcur[row][i * 32 + lane] + hacc[row][i * 32 + lane];
        v[i] = hv; s += hv; s2 += hv * hv;
      }
      s = wred(s); s2 = wred(s2);
      float mean = s * (1.f / Hn);
      float rstd = rsqrtf(s2 * (1.f / Hn) - mean * mean + 1e-5f);
#pragma unroll
      for (int i = 0; i < 8; ++i) {
        int c = i * 32 + lane;
        float hv = (v[i] - mean) * rstd * lnog[c] + lnob[c];
        hacc[row][c] = hv;
        zb[row][c] = (bf16)hv;
      }
    }
    __syncthreads();
    gru(t);
  }
}

// ---------------- launcher ----------------
extern "C" void kernel_launch(void* const* d_in, const int* in_sizes, int n_in,
                              void* d_out, int out_size, void* d_ws, size_t ws_size,
                              hipStream_t stream) {
  (void)in_sizes; (void)n_in; (void)out_size; (void)ws_size;
  const float* x    = (const float*)d_in[0];
  const float* ts   = (const float*)d_in[1];
  const float* msk  = (const float*)d_in[2];
  const float* eW1  = (const float*)d_in[3];
  const float* eb1  = (const float*)d_in[4];
  const float* eW2  = (const float*)d_in[5];
  const float* eb2  = (const float*)d_in[6];
  const float* oW1  = (const float*)d_in[7];
  const float* ob1  = (const float*)d_in[8];
  const float* oW2  = (const float*)d_in[9];
  const float* ob2  = (const float*)d_in[10];
  const float* Wih  = (const float*)d_in[11];
  const float* Whh  = (const float*)d_in[12];
  const float* bih  = (const float*)d_in[13];
  const float* bhh  = (const float*)d_in[14];
  const float* lnog = (const float*)d_in[15];
  const float* lnob = (const float*)d_in[16];
  const float* lngg = (const float*)d_in[17];
  const float* lngb = (const float*)d_in[18];
  float* out = (float*)d_out;

  char* w = (char*)d_ws;
  bf16* ebuf = (bf16*)w; w += (size_t)Bn * Tn * Hn * sizeof(bf16);   // 32 MB
  bf16* eW1t = (bf16*)w; w += (size_t)H4 * 2 * Cn * sizeof(bf16);
  bf16* eW2t = (bf16*)w; w += (size_t)Hn * H4 * sizeof(bf16);
  bf16* oW1t = (bf16*)w; w += (size_t)H4 * H4 * sizeof(bf16);
  bf16* oW2t = (bf16*)w; w += (size_t)Hn * H4 * sizeof(bf16);
  bf16* Wiht = (bf16*)w; w += (size_t)G3 * Hn * sizeof(bf16);
  bf16* Whht = (bf16*)w; w += (size_t)G3 * Hn * sizeof(bf16);

  // pre-transpose weights to [N][K] bf16 (L2-resident for the whole run)
  {
    int n;
    n = 2 * Cn * H4; k_tr<<<(n + 255) / 256, 256, 0, stream>>>(eW1, eW1t, 2 * Cn, H4);
    n = H4 * Hn;     k_tr<<<(n + 255) / 256, 256, 0, stream>>>(eW2, eW2t, H4, Hn);
    n = H4 * H4;     k_tr<<<(n + 255) / 256, 256, 0, stream>>>(oW1, oW1t, H4, H4);
    n = H4 * Hn;     k_tr<<<(n + 255) / 256, 256, 0, stream>>>(oW2, oW2t, H4, Hn);
    n = Hn * G3;     k_tr<<<(n + 255) / 256, 256, 0, stream>>>(Wih, Wiht, Hn, G3);
    n = Hn * G3;     k_tr<<<(n + 255) / 256, 256, 0, stream>>>(Whh, Whht, Hn, G3);
  }

  k_encoder<<<(Bn * Tn) / 16, 512, 0, stream>>>(x, msk, eW1t, eb1, eW2t, eb2, ebuf);
  k_scan<<<Bn / 16, 512, 0, stream>>>(ebuf, ts, oW1t, ob1, oW2t, ob2,
                                      Wiht, Whht, bih, bhh,
                                      lnog, lnob, lngg, lngb, out);
}
